// GCN120_71511205478662
// MI455X (gfx1250) — matrix operand order repaired and checked by reading, not compiled, not run on hardware
//
#include <hip/hip_runtime.h>

#define N_NODES  200000
#define N_EDGES  3200000
#define N_GRAPHS 400

typedef __attribute__((ext_vector_type(16))) _Float16 v16h;
typedef __attribute__((ext_vector_type(8)))  float    v8f;

// ---------------------------------------------------------------- utilities
__global__ void k_zero(float* __restrict__ p, size_t n) {
  size_t i = (size_t)blockIdx.x * blockDim.x + threadIdx.x;
  size_t stride = (size_t)gridDim.x * blockDim.x;
  for (; i < n; i += stride) p[i] = 0.f;
}

// deg[dst] += 1 per edge (f32 atomic, no-return)
__global__ void k_deg(const int* __restrict__ dst, float* __restrict__ deg, int nE) {
  int i = blockIdx.x * blockDim.x + threadIdx.x;
  int stride = gridDim.x * blockDim.x;
  for (; i < nE; i += stride) unsafeAtomicAdd(&deg[dst[i]], 1.f);
}

__global__ void k_invdeg(float* __restrict__ deg, int n) {
  int i = blockIdx.x * blockDim.x + threadIdx.x;
  if (i < n) deg[i] = 1.f / fmaxf(deg[i], 1.f);
}

// agg[dst,:] += h[src,:] ; one wave per edge, lanes stride features
__global__ void k_scatter(const float* __restrict__ h, const int* __restrict__ src,
                          const int* __restrict__ dst, float* __restrict__ agg,
                          int d, int nE) {
  int gw   = (blockIdx.x * blockDim.x + threadIdx.x) >> 5;
  int lane = threadIdx.x & 31;
  int nw   = (gridDim.x * blockDim.x) >> 5;
  for (int e = gw; e < nE; e += nw) {
    int s = src[e], t = dst[e];
    const float* __restrict__ hs = h   + (size_t)s * d;
    float* __restrict__       at = agg + (size_t)t * d;
    for (int j = lane; j < d; j += 32)
      unsafeAtomicAdd(&at[j], hs[j]);
  }
}

// ------------------------------------------------ fused SAGE projection (WMMA)
// out = [relu]( h @ Wself + (agg * inv_deg) @ Wneigh + b )
// One 256-thread block = 8 waves; each wave owns a 16-row M-tile and all
// N-tiles; K-loop in 32-wide f16 chunks via v_wmma_f32_16x16x32_f16.
template <int D_IN, int D_OUT, bool RELU>
__global__ __launch_bounds__(256) void k_sage_gemm(
    const float* __restrict__ h, const float* __restrict__ agg,
    const float* __restrict__ invd,
    const float* __restrict__ Wself, const float* __restrict__ Wneigh,
    const float* __restrict__ bias, float* __restrict__ out, int nNodes) {
  constexpr int KP  = ((D_IN + 31) / 32) * 32;   // padded K
  constexpr int NP  = ((D_OUT + 15) / 16) * 16;  // padded N
  constexpr int NKC = KP / 32;                   // K chunks
  constexpr int NNT = NP / 16;                   // N tiles

  // transposed f16 weights in LDS: sW[n * KP + k]
  __shared__ __align__(32) _Float16 sWs[NP * KP];
  __shared__ __align__(32) _Float16 sWn[NP * KP];

  // cooperative fill (NP*KP is a multiple of 256 for all layer shapes -> uniform)
  for (int idx = threadIdx.x; idx < NP * KP; idx += 256) {
    int n = idx / KP, k = idx % KP;
    bool ok = (n < D_OUT) & (k < D_IN);
    sWs[idx] = ok ? (_Float16)Wself[k * D_OUT + n]  : (_Float16)0.f;
    sWn[idx] = ok ? (_Float16)Wneigh[k * D_OUT + n] : (_Float16)0.f;
  }
  __syncthreads();

  const int wave  = threadIdx.x >> 5;
  const int lane  = threadIdx.x & 31;
  const int half  = lane >> 4;      // lane group 0/1
  const int nn    = lane & 15;
  const int mtile = blockIdx.x * 8 + wave;

  // A-matrix row for this lane (16-bit A layout: row = lane%16)
  const int  arow  = mtile * 16 + nn;
  const bool rowOK = arow < nNodes;
  const float scale = rowOK ? invd[arow] : 0.f;
  const float* __restrict__ hr = h   + (size_t)arow * D_IN;
  const float* __restrict__ ar = agg + (size_t)arow * D_IN;

  v8f acc[NNT];
#pragma unroll
  for (int t = 0; t < NNT; t++) acc[t] = v8f{0.f, 0.f, 0.f, 0.f, 0.f, 0.f, 0.f, 0.f};

#pragma unroll
  for (int kc = 0; kc < NKC; kc++) {
    // A fragment: lanes 0-15 -> K = kc*32 + 0..15; lanes 16-31 -> +16
    const int kbase = kc * 32 + half * 16;
    v16h ah, aa;
#pragma unroll
    for (int i = 0; i < 16; i++) {
      int k = kbase + i;
      bool ok = rowOK && (k < D_IN);
      ah[i] = (_Float16)(ok ? hr[k] : 0.f);
      aa[i] = (_Float16)(ok ? ar[k] * scale : 0.f);
    }
    // B fragment: lanes 0-15 -> K = kc*32 + 0..15; lanes 16-31 -> +16 (contig in Wt)
    const int kb = kc * 32 + half * 16;
#pragma unroll
    for (int t = 0; t < NNT; t++) {
      const v16h bs = *(const v16h*)&sWs[(t * 16 + nn) * KP + kb];
      const v16h bn = *(const v16h*)&sWn[(t * 16 + nn) * KP + kb];
      acc[t] = __builtin_amdgcn_wmma_f32_16x16x32_f16(false, ah, false, bs,
                                                      (short)0, acc[t], false, false);
      acc[t] = __builtin_amdgcn_wmma_f32_16x16x32_f16(false, aa, false, bn,
                                                      (short)0, acc[t], false, false);
    }
  }

  // C/D layout: lane group half, VGPR r -> row M = half*8 + r, col N = lane%16
#pragma unroll
  for (int t = 0; t < NNT; t++) {
    const int col = t * 16 + nn;
    if (col < D_OUT) {
      const float bv = bias[col];
#pragma unroll
      for (int r = 0; r < 8; r++) {
        const int rg = mtile * 16 + half * 8 + r;
        if (rg < nNodes) {
          float v = acc[t][r] + bv;
          if (RELU) v = fmaxf(v, 0.f);
          out[(size_t)rg * D_OUT + col] = v;
        }
      }
    }
  }
}

// ---------------------------------------------------------------- readout
__global__ void k_readout(const float* __restrict__ h, const int* __restrict__ gid,
                          float* __restrict__ tot, float* __restrict__ cnt, int n) {
  int i = blockIdx.x * blockDim.x + threadIdx.x;
  if (i >= n) return;
  int g = gid[i];
  unsafeAtomicAdd(&cnt[g], 1.f);
  const float* __restrict__ hr = h + (size_t)i * 5;
#pragma unroll
  for (int j = 0; j < 5; j++) unsafeAtomicAdd(&tot[g * 5 + j], hr[j]);
}

__global__ void k_final(const float* __restrict__ tot, const float* __restrict__ cnt,
                        float* __restrict__ out) {
  int i = blockIdx.x * blockDim.x + threadIdx.x;
  if (i < N_GRAPHS * 5) out[i] = tot[i] / fmaxf(cnt[i / 5], 1.f);
}

// ---------------------------------------------------------------- launch
extern "C" void kernel_launch(void* const* d_in, const int* in_sizes, int n_in,
                              void* d_out, int out_size, void* d_ws, size_t ws_size,
                              hipStream_t stream) {
  (void)in_sizes; (void)n_in; (void)out_size; (void)ws_size;
  const float* feat = (const float*)d_in[0];
  const int*   src  = (const int*)d_in[1];
  const int*   dst  = (const int*)d_in[2];
  const int*   gid  = (const int*)d_in[3];
  const float* Ws0 = (const float*)d_in[4],  *Wn0 = (const float*)d_in[5],  *b0 = (const float*)d_in[6];
  const float* Ws1 = (const float*)d_in[7],  *Wn1 = (const float*)d_in[8],  *b1 = (const float*)d_in[9];
  const float* Ws2 = (const float*)d_in[10], *Wn2 = (const float*)d_in[11], *b2 = (const float*)d_in[12];
  const float* Ws3 = (const float*)d_in[13], *Wn3 = (const float*)d_in[14], *b3 = (const float*)d_in[15];

  float* ws = (float*)d_ws;
  const size_t NB = (size_t)N_NODES * 128;
  float* bufA = ws;            // N x 128
  float* bufB = ws + NB;       // N x 128 (agg)
  float* bufC = ws + 2 * NB;   // N x 128
  float* invd = ws + 3 * NB;   // N
  float* tot  = invd + N_NODES;     // 400*5
  float* cnt  = tot + N_GRAPHS * 5; // 400

  const int gemmGrid = (N_NODES + 127) / 128;

  // degrees (shared by all layers)
  k_zero<<<1024, 256, 0, stream>>>(invd, (size_t)N_NODES);
  k_deg<<<4096, 256, 0, stream>>>(dst, invd, N_EDGES);
  k_invdeg<<<(N_NODES + 255) / 256, 256, 0, stream>>>(invd, N_NODES);

  // layer 0: feat(32) -> bufA(128)
  k_zero<<<4096, 256, 0, stream>>>(bufB, (size_t)N_NODES * 32);
  k_scatter<<<8192, 256, 0, stream>>>(feat, src, dst, bufB, 32, N_EDGES);
  k_sage_gemm<32, 128, true><<<gemmGrid, 256, 0, stream>>>(
      feat, bufB, invd, Ws0, Wn0, b0, bufA, N_NODES);

  // layer 1: bufA(128) -> bufC(48)
  k_zero<<<4096, 256, 0, stream>>>(bufB, (size_t)N_NODES * 128);
  k_scatter<<<8192, 256, 0, stream>>>(bufA, src, dst, bufB, 128, N_EDGES);
  k_sage_gemm<128, 48, true><<<gemmGrid, 256, 0, stream>>>(
      bufA, bufB, invd, Ws1, Wn1, b1, bufC, N_NODES);

  // layer 2: bufC(48) -> bufA(28)
  k_zero<<<4096, 256, 0, stream>>>(bufB, (size_t)N_NODES * 48);
  k_scatter<<<8192, 256, 0, stream>>>(bufC, src, dst, bufB, 48, N_EDGES);
  k_sage_gemm<48, 28, true><<<gemmGrid, 256, 0, stream>>>(
      bufC, bufB, invd, Ws2, Wn2, b2, bufA, N_NODES);

  // layer 3 (no relu): bufA(28) -> bufC(5)
  k_zero<<<4096, 256, 0, stream>>>(bufB, (size_t)N_NODES * 28);
  k_scatter<<<8192, 256, 0, stream>>>(bufA, src, dst, bufB, 28, N_EDGES);
  k_sage_gemm<28, 5, false><<<gemmGrid, 256, 0, stream>>>(
      bufA, bufB, invd, Ws3, Wn3, b3, bufC, N_NODES);

  // per-graph mean readout
  k_zero<<<8, 256, 0, stream>>>(tot, (size_t)(N_GRAPHS * 5 + N_GRAPHS));
  k_readout<<<(N_NODES + 255) / 256, 256, 0, stream>>>(bufC, gid, tot, cnt, N_NODES);
  k_final<<<(N_GRAPHS * 5 + 255) / 256, 256, 0, stream>>>(tot, cnt, (float*)d_out);
}
